// RecurrentDecoder_40776419508603
// MI455X (gfx1250) — compile-verified, matrix operand order checked by reading, not run on hardware
//
#include <hip/hip_runtime.h>
#include <hip/hip_bf16.h>
#include <math.h>

// ---------------------------------------------------------------------------
// Problem constants
// ---------------------------------------------------------------------------
#define VOCAB 32000
#define DIM   1024
#define HID   1024
#define SEQ_S 128
#define BATCH 32
#define SEQ_T 64
#define G4H   4096
#define BH    (BATCH * HID)   // 32768

// d_out layout (floats): logits (T,B,V) | attw (B,T,S) | hT (2,B,H) | cT | attnT
#define OUT_LOGITS 0
#define OUT_ATTW   ((size_t)SEQ_T * BATCH * VOCAB)
#define OUT_HT     (OUT_ATTW + (size_t)BATCH * SEQ_T * SEQ_S)
#define OUT_CT     (OUT_HT + (size_t)2 * BH)
#define OUT_ATTNT  (OUT_CT + (size_t)2 * BH)

// ---- workspace: f32 region then bf16 region (element offsets) -------------
#define WS_PRE0   ((size_t)0)                            // 2048 x 4096 f32
#define WS_G0     (WS_PRE0 + (size_t)2048 * 4096)        // 32 x 4096
#define WS_G1     (WS_G0 + (size_t)32 * 4096)
#define WS_APRE   (WS_G1 + (size_t)32 * 4096)            // 32 x 1024
#define WS_H0     (WS_APRE + (size_t)BH)                 // 2 buffers each
#define WS_C0     (WS_H0 + (size_t)2 * BH)
#define WS_H1     (WS_C0 + (size_t)2 * BH)
#define WS_C1     (WS_H1 + (size_t)2 * BH)
#define WS_ATTN   (WS_C1 + (size_t)2 * BH)
#define WS_B0SUM  (WS_ATTN + (size_t)2 * BH)             // 4096
#define WS_B1SUM  (WS_B0SUM + (size_t)4096)
#define WS_F32_END (WS_B1SUM + (size_t)4096)

// bf16 region (element offsets from bf16 base)
#define BW_WIH0   ((size_t)0)                            // 4096 x 2048
#define BW_WHH0   (BW_WIH0  + (size_t)4096 * 2048)       // 4096 x 1024
#define BW_WIH1   (BW_WHH0  + (size_t)4096 * 1024)
#define BW_WHH1   (BW_WIH1  + (size_t)4096 * 1024)
#define BW_WATTN  (BW_WHH1  + (size_t)4096 * 1024)       // 1024 x 2048
#define BW_WOUT   (BW_WATTN + (size_t)1024 * 2048)       // 32000 x 1024
#define BW_EMBSEQ (BW_WOUT  + (size_t)VOCAB * 1024)      // 2048 x 1024
#define BW_ATTNST (BW_EMBSEQ+ (size_t)2048 * 1024)       // 2048 x 1024
#define BW_H0BF   (BW_ATTNST+ (size_t)2048 * 1024)       // 2 buffers
#define BW_H1BF   (BW_H0BF  + (size_t)2 * BH)
#define BW_ABF    (BW_H1BF  + (size_t)2 * BH)
#define BW_CTX    (BW_ABF   + (size_t)2 * BH)            // 32 x 1024

typedef __attribute__((ext_vector_type(16))) __bf16 v16bf;
typedef __attribute__((ext_vector_type(8)))  float  v8f;
typedef __attribute__((ext_vector_type(4)))  unsigned int uint32x4;
typedef __attribute__((ext_vector_type(8)))  int int32x8;
typedef __attribute__((ext_vector_type(4)))  int int32x4;

__device__ __forceinline__ float sigf(float x) { return 1.0f / (1.0f + __expf(-x)); }

// ---------------------------------------------------------------------------
// TDM 2D tile load: global(bf16, row-major, stride in elems) -> LDS (packed).
// D# per CDNA5 ISA ch.8: group0 = {count|flags, lds_addr, gaddr_lo, gaddr_hi|type},
// group1 carries data_size/dims/tile/strides. Tiles are always in-bounds, so
// tensor dims are set huge to make OOB checks a no-op. data_size=1 -> 2 bytes.
// Must be called wave-uniformly; completion via TENSORcnt (in order per wave).
// 6-arg builtin on this toolchain: (g0, g1, g2, g3, g4, cpol).
// ---------------------------------------------------------------------------
__device__ __forceinline__ void tdm_load_2d(unsigned lds_byte_off, const __bf16* gptr,
                                            unsigned tile_k, unsigned tile_rows,
                                            unsigned long long stride_elems) {
  unsigned long long ga = (unsigned long long)(size_t)gptr;
  uint32x4 g0;
  g0[0] = 1u;                                            // count=1, user mode
  g0[1] = lds_byte_off;                                  // LDS byte address
  g0[2] = (unsigned)(ga & 0xffffffffu);                  // global_addr[31:0]
  g0[3] = (unsigned)((ga >> 32) & 0x1ffffffu) | (2u << 30); // addr[56:32] | type=2
  const unsigned td0 = 0x7fffffffu, td1 = 0x7fffffffu;   // huge tensor dims
  int32x8 g1;
  g1[0] = 0x00010000;                                    // data_size=1 (2B), mask=0
  g1[1] = (int)((td0 & 0xffffu) << 16);                  // tensor_dim0 lo
  g1[2] = (int)((td0 >> 16) | ((td1 & 0xffffu) << 16));  // dim0 hi | dim1 lo
  g1[3] = (int)((td1 >> 16) | (tile_k << 16));           // dim1 hi | tile_dim0
  g1[4] = (int)(tile_rows);                              // tile_dim1 | tile_dim2=0
  g1[5] = (int)(stride_elems & 0xffffffffu);             // dim0_stride lo
  g1[6] = (int)((stride_elems >> 32) & 0xffffu);         // stride hi | dim1_stride=0
  g1[7] = 0;
  int32x4 z4 = {0, 0, 0, 0};
  int32x8 z8 = {0, 0, 0, 0, 0, 0, 0, 0};
  __builtin_amdgcn_tensor_load_to_lds(g0, g1, z4, z4, z8, 0);
}

// ---------------------------------------------------------------------------
// WMMA GEMM: C[M,N] = A[M,K](bf16) * Bw[N,K](bf16)^T (+bias[N]) (+Cin[M,N])
// 256 threads = 8 waves; block tile 16 x 128 of C; one 16x16 C tile per wave.
// K chunk = 64. Double-buffered TDM pipeline: wave0 streams the 128x64 B tile,
// wave1 the 16x64 A tile, into LDS buffer (it+1)&1 while all waves compute
// from buffer it&1. s_wait_tensorcnt<=1 => current chunk landed (TDM ops
// complete in order per wave). Trailing barrier protects the buffer that the
// next iteration's TDM will overwrite. Fragments per ISA 7.12.2 layout.
// Single __shared__ block => static LDS offset 0 (TDM lds_addr assumption).
// M%16==0, N%128==0, K%64==0.
// ---------------------------------------------------------------------------
#define BUF_ELEMS  (16 * 64 + 128 * 64)   // 9216 bf16 = 18 KB per buffer
#define BUF_BYTES  (BUF_ELEMS * 2)

template <bool HAS_BIAS, bool HAS_CIN>
__global__ __launch_bounds__(256)
void gemm_wmma_tdm(const __bf16* __restrict__ A, int lda,
                   const __bf16* __restrict__ Bw, int ldb,
                   const float* __restrict__ bias,
                   const float* __restrict__ Cin,
                   float* __restrict__ C, int ldc, int K) {
  __shared__ __bf16 smem[2 * BUF_ELEMS];        // double-buffered: sA @0, sB @2048B

  const int tid  = threadIdx.x;
  const int w    = tid >> 5;                    // wave id (wave32)
  const int lane = tid & 31;
  const int m0   = blockIdx.y * 16;
  const int n0b  = blockIdx.x * 128;
  const int n0w  = n0b + w * 16;

  const int hh = lane >> 4;                     // lane half
  const int r  = lane & 15;
  const int kb = hh * 8;

  v8f acc = {};
  const int nIter = K >> 6;

  // prologue: chunk 0 into buffer 0
  if (w == 0)      tdm_load_2d(2048u, Bw + (size_t)n0b * ldb, 64u, 128u,
                               (unsigned long long)ldb);
  else if (w == 1) tdm_load_2d(0u,    A  + (size_t)m0  * lda, 64u, 16u,
                               (unsigned long long)lda);

  for (int it = 0; it < nIter; ++it) {
    const int cur = it & 1;
    if (it + 1 < nIter) {
      // issue next chunk into the other buffer (freed by previous iteration's
      // trailing barrier), then wait for the current chunk (<=1 outstanding).
      const unsigned nb = cur ? 0u : (unsigned)BUF_BYTES;
      const int kkn = (it + 1) << 6;
      if (w == 0)      tdm_load_2d(nb + 2048u, Bw + (size_t)n0b * ldb + kkn,
                                   64u, 128u, (unsigned long long)ldb);
      else if (w == 1) tdm_load_2d(nb, A + (size_t)m0 * lda + kkn,
                                   64u, 16u, (unsigned long long)lda);
      __builtin_amdgcn_s_wait_tensorcnt(1);
    } else {
      __builtin_amdgcn_s_wait_tensorcnt(0);
    }
    __syncthreads();                            // current tiles visible to all

    const __bf16* sA = smem + (cur ? BUF_ELEMS : 0);   // [16][64]
    const __bf16* sB = sA + 1024;                      // [128][64]
    #pragma unroll
    for (int g = 0; g < 2; ++g) {               // two 32-K WMMAs per chunk
      v16bf af, bf;
      #pragma unroll
      for (int j = 0; j < 16; ++j) {
        // 16-bit A-matrix 16x32 layout: lanes 0-15 K in {kb..kb+7,kb+16..kb+23}
        int kidx = (j < 8) ? (kb + j) : (kb + 8 + j);
        af[j] = sA[r * 64 + g * 32 + kidx];
        bf[j] = sB[(w * 16 + r) * 64 + g * 32 + kidx];
      }
      acc = __builtin_amdgcn_wmma_f32_16x16x32_bf16(
          false, af, false, bf, (short)0, acc, false, false);
    }
    __syncthreads();                            // buffer free for next TDM issue
  }

  // C layout: VGPR rr -> lanes 0-15: (M=rr, N=lane); lanes 16-31: (M=rr+8, ..)
  const int col = n0w + r;
  float bv = 0.0f;
  if (HAS_BIAS) bv = bias[col];
  #pragma unroll
  for (int rr = 0; rr < 8; ++rr) {
    int row = m0 + rr + hh * 8;
    float v = acc[rr] + bv;
    if (HAS_CIN) v += Cin[(size_t)row * ldc + col];
    C[(size_t)row * ldc + col] = v;
  }
}

// ---------------------------------------------------------------------------
// Elementwise / setup kernels
// ---------------------------------------------------------------------------
__global__ void f32_to_bf16(const float* __restrict__ in, __bf16* __restrict__ out,
                            size_t n) {
  size_t stride = (size_t)gridDim.x * 256;
  for (size_t i = blockIdx.x * 256ull + threadIdx.x; i < n; i += stride)
    out[i] = (__bf16)in[i];
}

__global__ void init_states(const float* __restrict__ h0, const float* __restrict__ c0,
                            const float* __restrict__ attn0,
                            float* h0b, float* c0b, float* h1b, float* c1b, float* ab,
                            __bf16* h0bf, __bf16* h1bf, __bf16* abf) {
  int i = blockIdx.x * 256 + threadIdx.x;
  if (i < BH) {
    float a = h0[i], b = h0[BH + i];
    h0b[i] = a; h1b[i] = b; h0bf[i] = (__bf16)a; h1bf[i] = (__bf16)b;
    c0b[i] = c0[i]; c1b[i] = c0[BH + i];
    float v = attn0[i]; ab[i] = v; abf[i] = (__bf16)v;
  }
}

__global__ void bias_sums(const float* bi0, const float* bh0,
                          const float* bi1, const float* bh1,
                          float* b0s, float* b1s) {
  int i = blockIdx.x * 256 + threadIdx.x;
  if (i < G4H) { b0s[i] = bi0[i] + bh0[i]; b1s[i] = bi1[i] + bh1[i]; }
}

__global__ void embed_gather(const int* __restrict__ words, const float* __restrict__ emb,
                             __bf16* __restrict__ emb_seq) {
  int row = blockIdx.x;                           // t*B + b
  int wid = words[row];
  const float* src = emb + (size_t)wid * DIM;
  __bf16* dst = emb_seq + (size_t)row * DIM;
  for (int d = threadIdx.x; d < DIM; d += 256) dst[d] = (__bf16)src[d];
}

__global__ void lstm_cell(const float* __restrict__ gates, const float* __restrict__ c_prev,
                          float* __restrict__ h_new, float* __restrict__ c_new,
                          __bf16* __restrict__ h_new_bf) {
  int idx = blockIdx.x * 256 + threadIdx.x;
  if (idx >= BH) return;
  int b = idx >> 10, j = idx & (HID - 1);
  const float* g = gates + (size_t)b * G4H;
  float gi = g[j], gf = g[HID + j], gg = g[2 * HID + j], go = g[3 * HID + j];
  float cn = sigf(gf) * c_prev[idx] + sigf(gi) * tanhf(gg);
  float hn = sigf(go) * tanhf(cn);
  c_new[idx] = cn; h_new[idx] = hn; h_new_bf[idx] = (__bf16)hn;
}

__global__ void attn_tanh(const float* __restrict__ pre, float* __restrict__ attn_f,
                          __bf16* __restrict__ attn_bf, __bf16* __restrict__ attn_state_t) {
  int idx = blockIdx.x * 256 + threadIdx.x;
  if (idx < BH) {
    float v = tanhf(pre[idx]);
    __bf16 vb = (__bf16)v;
    attn_f[idx] = v; attn_bf[idx] = vb; attn_state_t[idx] = vb;
  }
}

// scores -> masked softmax -> context vector; one block per batch element
__global__ __launch_bounds__(256)
void attn_scores(const float* __restrict__ context, const int* __restrict__ src_len,
                 const float* __restrict__ h1, __bf16* __restrict__ ctx_bf,
                 float* __restrict__ attw, int t) {
  const int b = blockIdx.x, tid = threadIdx.x;
  __shared__ float sout[HID];
  __shared__ float part[256];
  __shared__ float wv[SEQ_S];
  __shared__ float red[SEQ_S];

  for (int h = tid; h < HID; h += 256) sout[h] = h1[(size_t)b * HID + h];
  __syncthreads();
  {
    int s = tid >> 1, hs = (tid & 1) * 512;
    const float* cs = context + ((size_t)s * BATCH + b) * HID + hs;
    float acc = 0.0f;
    for (int h = 0; h < 512; ++h) acc += cs[h] * sout[hs + h];
    part[tid] = acc;
  }
  __syncthreads();
  int len = src_len[b];
  if (tid < SEQ_S) {
    float sc = part[2 * tid] + part[2 * tid + 1];
    if (tid >= len) sc = -1e9f;
    wv[tid] = sc; red[tid] = sc;
  }
  __syncthreads();
  for (int off = 64; off >= 1; off >>= 1) {
    if (tid < off) red[tid] = fmaxf(red[tid], red[tid + off]);
    __syncthreads();
  }
  float mx = red[0];
  __syncthreads();
  if (tid < SEQ_S) { float e = __expf(wv[tid] - mx); wv[tid] = e; red[tid] = e; }
  __syncthreads();
  for (int off = 64; off >= 1; off >>= 1) {
    if (tid < off) red[tid] += red[tid + off];
    __syncthreads();
  }
  float inv = 1.0f / red[0];
  __syncthreads();
  if (tid < SEQ_S) {
    float p = wv[tid] * inv;
    wv[tid] = p;
    attw[((size_t)b * SEQ_T + t) * SEQ_S + tid] = p;
  }
  __syncthreads();
  for (int h = tid; h < HID; h += 256) {
    float acc = 0.0f;
    for (int s = 0; s < SEQ_S; ++s) acc += wv[s] * context[((size_t)s * BATCH + b) * HID + h];
    ctx_bf[(size_t)b * HID + h] = (__bf16)acc;
  }
}

__global__ void final_copy(const float* h0b, const float* h1b,
                           const float* c0b, const float* c1b,
                           const float* ab, float* out) {
  int i = blockIdx.x * 256 + threadIdx.x;
  if (i < BH) {
    out[OUT_HT + i] = h0b[i];  out[OUT_HT + BH + i] = h1b[i];
    out[OUT_CT + i] = c0b[i];  out[OUT_CT + BH + i] = c1b[i];
    out[OUT_ATTNT + i] = ab[i];
  }
}

// ---------------------------------------------------------------------------
// Host orchestration
// ---------------------------------------------------------------------------
extern "C" void kernel_launch(void* const* d_in, const int* in_sizes, int n_in,
                              void* d_out, int out_size, void* d_ws, size_t ws_size,
                              hipStream_t stream) {
  const float* context = (const float*)d_in[0];
  const int*   src_len = (const int*)  d_in[1];
  const int*   words   = (const int*)  d_in[2];
  const float* h0_in   = (const float*)d_in[3];
  const float* c0_in   = (const float*)d_in[4];
  const float* attn0   = (const float*)d_in[5];
  const float* emb     = (const float*)d_in[6];
  const float* W_ih0   = (const float*)d_in[7];
  const float* W_hh0   = (const float*)d_in[8];
  const float* b_ih0   = (const float*)d_in[9];
  const float* b_hh0   = (const float*)d_in[10];
  const float* W_ih1   = (const float*)d_in[11];
  const float* W_hh1   = (const float*)d_in[12];
  const float* b_ih1   = (const float*)d_in[13];
  const float* b_hh1   = (const float*)d_in[14];
  const float* W_attn  = (const float*)d_in[15];
  const float* b_attn  = (const float*)d_in[16];
  const float* W_out   = (const float*)d_in[17];
  const float* b_out   = (const float*)d_in[18];

  float*  out = (float*)d_out;
  float*  ws  = (float*)d_ws;
  __bf16* bw  = (__bf16*)(ws + WS_F32_END);

  float* pre0   = ws + WS_PRE0;
  float* g0     = ws + WS_G0;
  float* g1     = ws + WS_G1;
  float* apre   = ws + WS_APRE;
  float* h0b[2] = { ws + WS_H0, ws + WS_H0 + BH };
  float* c0b[2] = { ws + WS_C0, ws + WS_C0 + BH };
  float* h1b[2] = { ws + WS_H1, ws + WS_H1 + BH };
  float* c1b[2] = { ws + WS_C1, ws + WS_C1 + BH };
  float* ab[2]  = { ws + WS_ATTN, ws + WS_ATTN + BH };
  float* b0s    = ws + WS_B0SUM;
  float* b1s    = ws + WS_B1SUM;

  __bf16* wih0  = bw + BW_WIH0;
  __bf16* whh0  = bw + BW_WHH0;
  __bf16* wih1  = bw + BW_WIH1;
  __bf16* whh1  = bw + BW_WHH1;
  __bf16* wattn = bw + BW_WATTN;
  __bf16* wout  = bw + BW_WOUT;
  __bf16* embsq = bw + BW_EMBSEQ;
  __bf16* attst = bw + BW_ATTNST;
  __bf16* h0bf[2] = { bw + BW_H0BF, bw + BW_H0BF + BH };
  __bf16* h1bf[2] = { bw + BW_H1BF, bw + BW_H1BF + BH };
  __bf16* abf[2]  = { bw + BW_ABF,  bw + BW_ABF  + BH };
  __bf16* ctxbf   = bw + BW_CTX;

  auto gemm = [&](const __bf16* A, int lda, const __bf16* B, int ldb,
                  const float* bias, const float* Cin, float* C, int ldc,
                  int M, int N, int K) {
    dim3 grid(N / 128, M / 16);
    if (bias)
      gemm_wmma_tdm<true, false><<<grid, 256, 0, stream>>>(A, lda, B, ldb, bias, Cin, C, ldc, K);
    else
      gemm_wmma_tdm<false, true><<<grid, 256, 0, stream>>>(A, lda, B, ldb, bias, Cin, C, ldc, K);
  };

  // ---- setup: states, bias sums, weight f32->bf16, embedding gather -------
  init_states<<<128, 256, 0, stream>>>(h0_in, c0_in, attn0,
                                       h0b[0], c0b[0], h1b[0], c1b[0], ab[0],
                                       h0bf[0], h1bf[0], abf[0]);
  bias_sums<<<16, 256, 0, stream>>>(b_ih0, b_hh0, b_ih1, b_hh1, b0s, b1s);
  f32_to_bf16<<<2048, 256, 0, stream>>>(W_ih0,  wih0,  (size_t)4096 * 2048);
  f32_to_bf16<<<2048, 256, 0, stream>>>(W_hh0,  whh0,  (size_t)4096 * 1024);
  f32_to_bf16<<<2048, 256, 0, stream>>>(W_ih1,  wih1,  (size_t)4096 * 1024);
  f32_to_bf16<<<2048, 256, 0, stream>>>(W_hh1,  whh1,  (size_t)4096 * 1024);
  f32_to_bf16<<<2048, 256, 0, stream>>>(W_attn, wattn, (size_t)1024 * 2048);
  f32_to_bf16<<<4096, 256, 0, stream>>>(W_out,  wout,  (size_t)VOCAB * 1024);
  embed_gather<<<SEQ_T * BATCH, 256, 0, stream>>>(words, emb, embsq);

  // pre0 = emb_seq @ W_ih0[:, :D]^T + (b_ih0 + b_hh0)  (parallel over all T)
  gemm(embsq, DIM, wih0, 2048, b0s, nullptr, pre0, G4H, SEQ_T * BATCH, G4H, DIM);

  // ---- sequential decode ---------------------------------------------------
  for (int t = 0; t < SEQ_T; ++t) {
    int r = t & 1, wnew = (t + 1) & 1;
    const float* pre0_t = pre0 + (size_t)t * BATCH * G4H;

    gemm(abf[r],  HID, wih0 + DIM, 2048, nullptr, pre0_t, g0, G4H, BATCH, G4H, HID);
    gemm(h0bf[r], HID, whh0, 1024, nullptr, g0, g0, G4H, BATCH, G4H, HID);
    lstm_cell<<<128, 256, 0, stream>>>(g0, c0b[r], h0b[wnew], c0b[wnew], h0bf[wnew]);

    gemm(h0bf[wnew], HID, wih1, 1024, b1s, nullptr, g1, G4H, BATCH, G4H, HID);
    gemm(h1bf[r],    HID, whh1, 1024, nullptr, g1, g1, G4H, BATCH, G4H, HID);
    lstm_cell<<<128, 256, 0, stream>>>(g1, c1b[r], h1b[wnew], c1b[wnew], h1bf[wnew]);

    attn_scores<<<BATCH, 256, 0, stream>>>(context, src_len, h1b[wnew], ctxbf,
                                           out + OUT_ATTW, t);
    gemm(ctxbf,      HID, wattn,       2048, b_attn, nullptr, apre, HID, BATCH, HID, HID);
    gemm(h1bf[wnew], HID, wattn + HID, 2048, nullptr, apre,    apre, HID, BATCH, HID, HID);
    attn_tanh<<<128, 256, 0, stream>>>(apre, ab[wnew], abf[wnew],
                                       attst + (size_t)t * BH);
  }

  // logits = attn_states @ W_out^T + b_out   (M=2048, N=32000, K=1024)
  gemm(attst, HID, wout, 1024, b_out, nullptr, out + OUT_LOGITS, VOCAB,
       SEQ_T * BATCH, VOCAB, HID);

  final_copy<<<128, 256, 0, stream>>>(h0b[0], h1b[0], c0b[0], c1b[0], ab[0], out);
}
